// GATEncoder_47321949667561
// MI455X (gfx1250) — matrix operand order repaired and checked by reading, not compiled, not run on hardware
//
#include <hip/hip_runtime.h>
#include <hip/hip_bf16.h>
#include <stdint.h>

typedef __attribute__((ext_vector_type(16))) __bf16 v16bf;
typedef __attribute__((ext_vector_type(8)))  __bf16 v8bf;
typedef __attribute__((ext_vector_type(8)))  float  v8f;
typedef __attribute__((ext_vector_type(4)))  int    v4i;
typedef __attribute__((address_space(1))) v4i* gv4i_p;   // global 16B vector ptr
typedef __attribute__((address_space(3))) v4i* lv4i_p;   // LDS    16B vector ptr

union Frag16 { v16bf v; v8bf h[2]; };

#if __has_builtin(__builtin_amdgcn_global_load_async_to_lds_b128)
#define GAT_HAVE_ASYNC 1
#endif

static __device__ __forceinline__ unsigned short f32_to_bf16(float f) {
    unsigned u = __float_as_uint(f);
    unsigned r = u + 0x7fffu + ((u >> 16) & 1u);   // round-to-nearest-even
    return (unsigned short)(r >> 16);
}
static __device__ __forceinline__ int   fenc(float f) { int i = __float_as_int(f); return i >= 0 ? i : (i ^ 0x7fffffff); }
static __device__ __forceinline__ float fdec(int i)   { return __int_as_float(i >= 0 ? i : (i ^ 0x7fffffff)); }

// ---------------------------------------------------------------- utilities
__global__ void gat_fill_u32(uint32_t* __restrict__ p, uint32_t v, size_t n) {
    size_t t = (size_t)blockIdx.x * blockDim.x + threadIdx.x;
    if (t < n) p[t] = v;
}
__global__ void gat_cast_bf16(const float* __restrict__ s, unsigned short* __restrict__ d, size_t n) {
    size_t t = (size_t)blockIdx.x * blockDim.x + threadIdx.x;
    if (t < n) d[t] = f32_to_bf16(s[t]);
}
// W[K,Nn] (row-major) -> Wt[Nn,K] bf16 so B-fragment loads are contiguous per column
__global__ void gat_transpose_cast(const float* __restrict__ W, unsigned short* __restrict__ Wt,
                                   int K, int Nn) {
    size_t t = (size_t)blockIdx.x * blockDim.x + threadIdx.x;
    if (t >= (size_t)K * Nn) return;
    int k = (int)(t / Nn), n = (int)(t % Nn);
    Wt[(size_t)n * K + k] = f32_to_bf16(W[t]);
}

// ---------------------------------------------------------------- WMMA GEMM
// C[M,Nn] f32 = A[M,K] bf16 * Bt[Nn,K] bf16   (K%32==0, Nn%128==0; M guarded)
// Block: 256 threads = 8 waves; block tile = 64(M) x 128(N).
// Each wave owns a 16-wide N strip and computes 4 stacked 16x16 tiles
// (4 WMMAs / K-step, B fragment reused 4x in registers).
// A tile (64x32 bf16) double-buffered in LDS, filled with async LDS loads.
__global__ __launch_bounds__(256) void gat_wmma_gemm(
    const unsigned short* __restrict__ A,
    const unsigned short* __restrict__ Bt,
    float* __restrict__ C, int M, int K, int Nn)
{
    __shared__ unsigned short As[2][64 * 32];   // 2 x 4KB double buffer
    const int tid  = threadIdx.x;
    const int wid  = tid >> 5;
    const int lane = tid & 31;
    const int r15  = lane & 15;                 // A-frag row / B-frag col / C col
    const int kg   = lane >> 4;                 // half-wave selector
    const size_t rowbase = (size_t)blockIdx.x * 64;
    const int colg = blockIdx.y * 128 + wid * 16 + r15;
    const unsigned short* brow = Bt + (size_t)colg * K;

    // cooperative A-tile producer: 64x32 bf16 = 2048 ushorts; 16B per thread
    const int f  = tid * 8;
    const int rr = f >> 5;
    const int cc = f & 31;

    v8f acc0 = {}, acc1 = {}, acc2 = {}, acc3 = {};

    // prologue: stage first tile
    {
        const unsigned short* g = &A[(rowbase + rr) * (size_t)K + cc];
#ifdef GAT_HAVE_ASYNC
        __builtin_amdgcn_global_load_async_to_lds_b128(
            (gv4i_p)g, (lv4i_p)&As[0][f], 0, 0);
        asm volatile("s_wait_asynccnt 0x0" ::: "memory");
#else
        *(v8bf*)&As[0][f] = *(const v8bf*)g;
#endif
    }
    __syncthreads();

    int cur = 0;
    for (int kb = 0; kb < K; kb += 32) {
        if (kb + 32 < K) {                       // stage next tile into other buffer
            const unsigned short* g = &A[(rowbase + rr) * (size_t)K + kb + 32 + cc];
#ifdef GAT_HAVE_ASYNC
            __builtin_amdgcn_global_load_async_to_lds_b128(
                (gv4i_p)g, (lv4i_p)&As[cur ^ 1][f], 0, 0);
#else
            *(v8bf*)&As[cur ^ 1][f] = *(const v8bf*)g;
#endif
        }

        // B fragment (contiguous-k order, same as A): half-wave holds column r15, contiguous 16 K
        Frag16 fb;
        const v8bf* bp = (const v8bf*)(brow + kb + kg * 16);
        fb.h[0] = bp[0];
        fb.h[1] = bp[1];

        // 4 stacked A fragments from LDS (contiguous-k order: k = 16*kg + s)
        #pragma unroll
        for (int i = 0; i < 4; ++i) {
            Frag16 fa;
            const unsigned short* ab = &As[cur][(i * 16 + r15) * 32];
            fa.h[0] = *(const v8bf*)&ab[kg * 16];
            fa.h[1] = *(const v8bf*)&ab[kg * 16 + 8];
            v8f& acc = (i == 0) ? acc0 : (i == 1) ? acc1 : (i == 2) ? acc2 : acc3;
            acc = __builtin_amdgcn_wmma_f32_16x16x32_bf16(false, fa.v, false, fb.v,
                                                          (short)0, acc, false, false);
        }

#ifdef GAT_HAVE_ASYNC
        asm volatile("s_wait_asynccnt 0x0" ::: "memory");
#endif
        __syncthreads();                         // one barrier per K-step
        cur ^= 1;
    }

    // C layout per tile: VGPR v -> M = v + 8*kg, N = lane&15 (bounds-guarded rows)
    #pragma unroll
    for (int i = 0; i < 4; ++i) {
        const v8f& acc = (i == 0) ? acc0 : (i == 1) ? acc1 : (i == 2) ? acc2 : acc3;
        int mbase = (int)rowbase + i * 16 + 8 * kg;
        #pragma unroll
        for (int v = 0; v < 8; ++v) {
            int m = mbase + v;
            if (m < M) C[(size_t)m * Nn + colg] = acc[v];
        }
    }
}

// ---------------------------------------------------------------- attention
// per-node attention logits: asrc[n,h] = <h[n,h,:], a_src[h,:]>  (C=64, wave per head)
__global__ __launch_bounds__(256) void gat_node_alpha(
    const float* __restrict__ h, const float* __restrict__ a_src,
    const float* __restrict__ a_dst, float* __restrict__ asrc, float* __restrict__ adst)
{
    int n = blockIdx.x, hd = threadIdx.x >> 5, lane = threadIdx.x & 31;
    const float* hp = h + (size_t)n * 512 + hd * 64;
    const float* as = a_src + hd * 64;
    const float* ad = a_dst + hd * 64;
    float v0 = hp[lane], v1 = hp[lane + 32];
    float s1 = v0 * as[lane] + v1 * as[lane + 32];
    float s2 = v0 * ad[lane] + v1 * ad[lane + 32];
    #pragma unroll
    for (int off = 16; off > 0; off >>= 1) {
        s1 += __shfl_down(s1, off, 32);
        s2 += __shfl_down(s2, off, 32);
    }
    if (lane == 0) { asrc[n * 8 + hd] = s1; adst[n * 8 + hd] = s2; }
}

static __device__ __forceinline__ void edge_ends(const int* __restrict__ ei, int E, int eid,
                                                 int& s, int& d) {
    if (eid < E) { s = ei[eid]; d = ei[E + eid]; }
    else         { s = d = eid - E; }            // self-loop
}

__global__ void gat_edge_max(const int* __restrict__ ei, const float* __restrict__ asrc,
                             const float* __restrict__ adst, float* __restrict__ ebuf,
                             int* __restrict__ menc, int E, int N)
{
    int t = blockIdx.x * blockDim.x + threadIdx.x;
    if (t >= (E + N) * 8) return;
    int eid = t >> 3, hd = t & 7, s, d;
    edge_ends(ei, E, eid, s, d);
    float e = asrc[s * 8 + hd] + adst[d * 8 + hd];
    e = e > 0.f ? e : 0.2f * e;                 // leaky relu
    ebuf[t] = e;
    atomicMax(&menc[d * 8 + hd], fenc(e));      // segment max (ordered-int float max)
}

__global__ void gat_edge_expsum(const int* __restrict__ ei, float* __restrict__ ebuf,
                                const int* __restrict__ menc, float* __restrict__ ssum,
                                int E, int N)
{
    int t = blockIdx.x * blockDim.x + threadIdx.x;
    if (t >= (E + N) * 8) return;
    int eid = t >> 3, hd = t & 7, s, d;
    edge_ends(ei, E, eid, s, d);
    float ex = __expf(ebuf[t] - fdec(menc[d * 8 + hd]));
    ebuf[t] = ex;
    atomicAdd(&ssum[d * 8 + hd], ex);
}

// message aggregation: wave per (edge, head); 32 lanes cover C=64
__global__ __launch_bounds__(256) void gat_edge_agg(
    const int* __restrict__ ei, const float* __restrict__ h,
    const float* __restrict__ ebuf, const float* __restrict__ ssum,
    float* __restrict__ agg, int E, int N)
{
    int eid = blockIdx.x, hd = threadIdx.x >> 5, lane = threadIdx.x & 31, s, d;
    edge_ends(ei, E, eid, s, d);
    float alpha = ebuf[eid * 8 + hd] / (ssum[d * 8 + hd] + 1e-16f);
    const float* hp = h + (size_t)s * 512 + hd * 64;
    float* ap = agg + (size_t)d * 512 + hd * 64;
    atomicAdd(&ap[lane],      hp[lane]      * alpha);
    atomicAdd(&ap[lane + 32], hp[lane + 32] * alpha);
}

// layer-1 epilogue: elu(agg + b1) -> bf16 input of layer 2
__global__ void gat_post1(const float* __restrict__ agg, const float* __restrict__ b,
                          unsigned short* __restrict__ hb, size_t total)
{
    size_t t = (size_t)blockIdx.x * blockDim.x + threadIdx.x;
    if (t >= total) return;
    float v = agg[t] + b[t & 511];
    v = v > 0.f ? v : __expf(v) - 1.f;
    hb[t] = f32_to_bf16(v);
}

// layer-2 epilogue: head-mean + b2, elu, segment-mean-pool accumulate
__global__ __launch_bounds__(64) void gat_post2(
    const float* __restrict__ agg, const float* __restrict__ b2,
    const int* __restrict__ batch, float* __restrict__ pooled, float* __restrict__ cnt)
{
    int n = blockIdx.x, c = threadIdx.x;
    const float* ap = agg + (size_t)n * 512;
    float acc = 0.f;
    #pragma unroll
    for (int hd = 0; hd < 8; ++hd) acc += ap[hd * 64 + c];
    float v = acc * 0.125f + b2[c];
    v = v > 0.f ? v : __expf(v) - 1.f;
    int g = batch[n];
    atomicAdd(&pooled[g * 64 + c], v);
    if (c == 0) atomicAdd(&cnt[g], 1.f);
}

__global__ void gat_pool_div(const float* __restrict__ pooled, const float* __restrict__ cnt,
                             float* __restrict__ out)
{
    int t = blockIdx.x * blockDim.x + threadIdx.x;
    if (t >= 64 * 64) return;
    out[t] = pooled[t] / fmaxf(cnt[t >> 6], 1.f);
}

// ---------------------------------------------------------------- launcher
extern "C" void kernel_launch(void* const* d_in, const int* in_sizes, int n_in,
                              void* d_out, int out_size, void* d_ws, size_t ws_size,
                              hipStream_t stream)
{
    const float* x   = (const float*)d_in[0];
    const int*   ei  = (const int*)  d_in[1];
    const int*   bat = (const int*)  d_in[2];
    const float* W1  = (const float*)d_in[3];
    const float* as1 = (const float*)d_in[4];
    const float* ad1 = (const float*)d_in[5];
    const float* b1  = (const float*)d_in[6];
    const float* W2  = (const float*)d_in[7];
    const float* as2 = (const float*)d_in[8];
    const float* ad2 = (const float*)d_in[9];
    const float* b2  = (const float*)d_in[10];
    float* out = (float*)d_out;
    (void)n_in; (void)out_size; (void)ws_size;

    const int F1 = 128, HC = 512;
    const int N  = in_sizes[0] / F1;            // 50000
    const int E  = in_sizes[1] / 2;             // 400000
    const int E2 = E + N;                       // + self loops

    auto au = [](size_t v) { return (v + 255) & ~(size_t)255; };
    char* p = (char*)d_ws;
    unsigned short* Xb   = (unsigned short*)p; p += au((size_t)N * F1 * 2);
    unsigned short* Wt   = (unsigned short*)p; p += au((size_t)512 * 512 * 2);
    float*          hbuf = (float*)p;          p += au((size_t)N * HC * 4);
    float*          agg  = (float*)p;          p += au((size_t)N * HC * 4);
    unsigned short* hb16 = (unsigned short*)p; p += au((size_t)N * HC * 2);
    float*          asrc = (float*)p;          p += au((size_t)N * 8 * 4);
    float*          adst = (float*)p;          p += au((size_t)N * 8 * 4);
    int*            menc = (int*)p;            p += au((size_t)N * 8 * 4);
    float*          ssum = (float*)p;          p += au((size_t)N * 8 * 4);
    float*          ebuf = (float*)p;          p += au((size_t)E2 * 8 * 4);
    float*          pooled = (float*)p;        p += au((size_t)64 * 64 * 4);
    float*          cnt  = (float*)p;

    dim3 blk(256);
    auto g1 = [](size_t n) { return dim3((unsigned)((n + 255) / 256)); };
    const unsigned mg = (unsigned)((N + 63) / 64);   // guarded M tiles of 64

    // ---- precision prep + init
    gat_cast_bf16<<<g1((size_t)N * F1), blk, 0, stream>>>(x, Xb, (size_t)N * F1);
    gat_transpose_cast<<<g1((size_t)F1 * HC), blk, 0, stream>>>(W1, Wt, F1, HC);
    gat_fill_u32<<<g1((size_t)N * 8), blk, 0, stream>>>((uint32_t*)menc, 0x80000000u, (size_t)N * 8);
    gat_fill_u32<<<g1((size_t)N * 8), blk, 0, stream>>>((uint32_t*)ssum, 0u, (size_t)N * 8);
    gat_fill_u32<<<g1((size_t)N * HC), blk, 0, stream>>>((uint32_t*)agg, 0u, (size_t)N * HC);
    gat_fill_u32<<<g1(64 * 64 + 64), blk, 0, stream>>>((uint32_t*)pooled, 0u, 64 * 64 + 64);

    // ---- layer 1
    gat_wmma_gemm<<<dim3(mg, HC / 128), blk, 0, stream>>>(Xb, Wt, hbuf, N, F1, HC);
    gat_node_alpha<<<dim3(N), blk, 0, stream>>>(hbuf, as1, ad1, asrc, adst);
    gat_edge_max<<<g1((size_t)E2 * 8), blk, 0, stream>>>(ei, asrc, adst, ebuf, menc, E, N);
    gat_edge_expsum<<<g1((size_t)E2 * 8), blk, 0, stream>>>(ei, ebuf, menc, ssum, E, N);
    gat_edge_agg<<<dim3(E2), blk, 0, stream>>>(ei, hbuf, ebuf, ssum, agg, E, N);
    gat_post1<<<g1((size_t)N * HC), blk, 0, stream>>>(agg, b1, hb16, (size_t)N * HC);

    // ---- layer 2
    gat_transpose_cast<<<g1((size_t)HC * HC), blk, 0, stream>>>(W2, Wt, HC, HC);
    gat_fill_u32<<<g1((size_t)N * 8), blk, 0, stream>>>((uint32_t*)menc, 0x80000000u, (size_t)N * 8);
    gat_fill_u32<<<g1((size_t)N * 8), blk, 0, stream>>>((uint32_t*)ssum, 0u, (size_t)N * 8);
    gat_fill_u32<<<g1((size_t)N * HC), blk, 0, stream>>>((uint32_t*)agg, 0u, (size_t)N * HC);
    gat_wmma_gemm<<<dim3(mg, HC / 128), blk, 0, stream>>>(hb16, Wt, hbuf, N, HC, HC);
    gat_node_alpha<<<dim3(N), blk, 0, stream>>>(hbuf, as2, ad2, asrc, adst);
    gat_edge_max<<<g1((size_t)E2 * 8), blk, 0, stream>>>(ei, asrc, adst, ebuf, menc, E, N);
    gat_edge_expsum<<<g1((size_t)E2 * 8), blk, 0, stream>>>(ei, ebuf, menc, ssum, E, N);
    gat_edge_agg<<<dim3(E2), blk, 0, stream>>>(ei, hbuf, ebuf, ssum, agg, E, N);
    gat_post2<<<dim3(N), dim3(64), 0, stream>>>(agg, b2, bat, pooled, cnt);
    gat_pool_div<<<g1(64 * 64), blk, 0, stream>>>(pooled, cnt, out);
}
